// RelPartialLearnableMultiHeadAttn_10419590660461
// MI455X (gfx1250) — matrix operand order repaired and checked by reading, not compiled, not run on hardware
//
#include <hip/hip_runtime.h>

#define S_LEN 1024
#define BSZ   8
#define NHD   16
#define DH    64
#define DM    1024
#define NDIM  1024
#define BH    (BSZ * NHD)
#define SB    (S_LEN * BSZ)

typedef __attribute__((ext_vector_type(16))) __bf16 v16bf;
typedef __attribute__((ext_vector_type(8)))  float  v8f;

union FragBF { uint4 u[2]; v16bf v; };
static_assert(sizeof(FragBF) == 32, "frag size");

__device__ __forceinline__ v8f wmma_bf16(v16bf a, v16bf b, v8f c) {
  // D(16x16,f32) = A(16x32,bf16) * B(32x16,bf16) + C
  return __builtin_amdgcn_wmma_f32_16x16x32_bf16(false, a, false, b, (short)0, c, false, false);
}

// ---- CDNA5 async memory->LDS copy (GVS mode), tracked by ASYNCcnt ----
__device__ __forceinline__ unsigned lds_addr32(const void* p) {
  // generic pointers to LDS carry the wave-relative LDS byte offset in the
  // low 32 bits (aperture bits live in [63:32])
  return (unsigned)(uintptr_t)p;
}
__device__ __forceinline__ void async_ld_b128(unsigned lds_off, unsigned gbyte_off,
                                              const void* sbase) {
  asm volatile("global_load_async_to_lds_b128 %0, %1, %2"
               :: "v"(lds_off), "v"(gbyte_off), "s"(sbase)
               : "memory");
}
__device__ __forceinline__ void wait_async0() {
  asm volatile("s_wait_asynccnt 0x0" ::: "memory");
}

// ---------------- f32 -> bf16 cast ----------------
__global__ void cast_f32_bf16_kernel(const float* __restrict__ src,
                                     __bf16* __restrict__ dst, int n) {
  int i = blockIdx.x * blockDim.x + threadIdx.x;
  int stride = gridDim.x * blockDim.x;
  for (; i < n; i += stride) dst[i] = (__bf16)src[i];
}

// ---------------- bf16 GEMM: C[M,N] = A[M,K] * B[K,N], row major ----------------
// Block tile 128x128, K-step 32, 256 threads = 8 waves laid 2(M) x 4(N);
// each wave owns 64x32 = 4x2 WMMA tiles. A tile staged via async-to-LDS.
template <bool OUT_BF16>
__global__ __launch_bounds__(256) void gemm_bf16_kernel(
    const __bf16* __restrict__ A, const __bf16* __restrict__ B,
    void* __restrict__ Cv, int M, int N, int K) {
  __shared__ __align__(16) __bf16 As[128][32];
  __shared__ __align__(16) __bf16 Bt[128][32];  // transposed: [n][k]

  const int tid  = threadIdx.x;
  const int lane = tid & 31, w = tid >> 5;
  const int lm = lane & 15, hi = lane >> 4;
  const int wm = w >> 2, wn = w & 3;
  const int rowBase = blockIdx.y * 128;
  const int colBase = blockIdx.x * 128;
  const unsigned asBase = lds_addr32(&As[0][0]);

  v8f acc[4][2];
#pragma unroll
  for (int mi = 0; mi < 4; ++mi)
#pragma unroll
    for (int ni = 0; ni < 2; ++ni) acc[mi][ni] = (v8f){0.f,0.f,0.f,0.f,0.f,0.f,0.f,0.f};

  for (int k0 = 0; k0 < K; k0 += 32) {
    __syncthreads();  // previous-iteration fragment reads done before refill
    // stage A tile 128x32: direct memory->LDS async copies (no VGPR round-trip)
    for (int c = tid; c < 512; c += 256) {
      int r = c >> 2, col8 = (c & 3) << 3;
      unsigned ldsoff = asBase + (unsigned)(r * 32 + col8) * 2u;
      unsigned goff   = ((unsigned)(rowBase + r) * (unsigned)K + (unsigned)(k0 + col8)) * 2u;
      async_ld_b128(ldsoff, goff, A);
    }
    // stage B tile 32x128, transposed into Bt[n][k] (transpose needs VGPRs)
    for (int c = tid; c < 512; c += 256) {
      int gr = c >> 4, gc = (c & 15) << 3;
      uint4 v = *(const uint4*)(&B[(size_t)(k0 + gr) * N + colBase + gc]);
      const __bf16* pv = reinterpret_cast<const __bf16*>(&v);
#pragma unroll
      for (int e = 0; e < 8; ++e) Bt[gc + e][gr] = pv[e];
    }
    if (k0 + 32 < K) {
      __builtin_prefetch(&A[(size_t)(rowBase + (tid >> 1)) * K + k0 + 32], 0, 1);
      __builtin_prefetch(&B[(size_t)(k0 + 32 + (tid >> 3)) * N + colBase + (tid & 7) * 16], 0, 1);
    }
    wait_async0();     // this wave's async LDS writes have landed
    __syncthreads();   // join all waves' fills

    FragBF af[4], bf_[2];
#pragma unroll
    for (int mi = 0; mi < 4; ++mi) {
      const __bf16* p = As[wm * 64 + mi * 16 + lm];
      af[mi].u[0] = *(const uint4*)(p + hi * 8);
      af[mi].u[1] = *(const uint4*)(p + 16 + hi * 8);
    }
#pragma unroll
    for (int ni = 0; ni < 2; ++ni) {
      const __bf16* p = Bt[wn * 32 + ni * 16 + lm];
      bf_[ni].u[0] = *(const uint4*)(p + hi * 8);
      bf_[ni].u[1] = *(const uint4*)(p + 16 + hi * 8);
    }
#pragma unroll
    for (int mi = 0; mi < 4; ++mi)
#pragma unroll
      for (int ni = 0; ni < 2; ++ni)
        acc[mi][ni] = wmma_bf16(af[mi].v, bf_[ni].v, acc[mi][ni]);
  }

#pragma unroll
  for (int mi = 0; mi < 4; ++mi)
#pragma unroll
    for (int ni = 0; ni < 2; ++ni) {
      int r0 = rowBase + wm * 64 + mi * 16 + hi * 8;
      int c0 = colBase + wn * 32 + ni * 16 + lm;
#pragma unroll
      for (int r = 0; r < 8; ++r) {
        float val = acc[mi][ni][r];
        if (OUT_BF16)
          ((__bf16*)Cv)[(size_t)(r0 + r) * N + c0] = (__bf16)val;
        else
          ((float*)Cv)[(size_t)(r0 + r) * N + c0] = val;
      }
    }
}

// ---------------- head layout kernels ----------------
__global__ void make_q_heads_kernel(const __bf16* __restrict__ Qb,
                                    const float* __restrict__ rwb,
                                    const float* __restrict__ rrb,
                                    __bf16* __restrict__ Qw,
                                    __bf16* __restrict__ Qr) {
  int idx = blockIdx.x * 256 + threadIdx.x;  // S*B*NHD*DH elements
  int d = idx & 63, n = (idx >> 6) & 15, b = (idx >> 10) & 7, s = idx >> 13;
  float q = (float)Qb[(size_t)(s * BSZ + b) * NDIM + n * DH + d];
  size_t o = ((size_t)(b * NHD + n) * S_LEN + s) * DH + d;
  Qw[o] = (__bf16)(q + rwb[n * DH + d]);
  Qr[o] = (__bf16)(q + rrb[n * DH + d]);
}

__global__ void make_kv_heads_kernel(const __bf16* __restrict__ KVb,
                                     __bf16* __restrict__ Kh,
                                     __bf16* __restrict__ Vt) {
  int idx = blockIdx.x * 256 + threadIdx.x;
  int d = idx & 63, n = (idx >> 6) & 15, b = (idx >> 10) & 7, s = idx >> 13;
  size_t base = (size_t)(s * BSZ + b) * (2 * NDIM);
  __bf16 kk = KVb[base + n * DH + d];
  __bf16 vv = KVb[base + NDIM + n * DH + d];
  Kh[((size_t)(b * NHD + n) * S_LEN + s) * DH + d] = kk;          // [bh][j][d]
  Vt[((size_t)(b * NHD + n) * DH + d) * S_LEN + s] = vv;          // [bh][d][j]
}

__global__ void make_r_heads_kernel(const __bf16* __restrict__ Rb,
                                    __bf16* __restrict__ Rh) {
  int idx = blockIdx.x * 256 + threadIdx.x;  // S*NHD*DH
  int d = idx & 63, n = (idx >> 6) & 15, s = idx >> 10;
  Rh[((size_t)n * S_LEN + s) * DH + d] = Rb[(size_t)s * NDIM + n * DH + d];
}

// ---------------- fused attention per (b, head, 64-query block) ----------------
// LDS: scores 64x1024 f32 (256KB, only possible with CDNA5's 320KB/WGP),
// qw 64x64 bf16, qr 80x64 bf16, reduce buffers.
__global__ __launch_bounds__(256) void attn_kernel(
    const __bf16* __restrict__ Qw, const __bf16* __restrict__ Qr,
    const __bf16* __restrict__ Kh, const __bf16* __restrict__ Vt,
    const __bf16* __restrict__ Rh, __bf16* __restrict__ AV) {
  extern __shared__ char smem[];
  float*  scores = (float*)smem;                                  // 64*1024
  __bf16* qw_s   = (__bf16*)(smem + 64 * 1024 * 4);               // 64*64
  __bf16* qr_s   = qw_s + 64 * 64;                                // 80*64
  float*  pbuf   = (float*)((char*)(qr_s + 80 * 64));             // 256
  float*  rowsum = pbuf + 256;                                    // 64

  const int iBlk = blockIdx.x, n = blockIdx.y, b = blockIdx.z;
  const int bh = b * NHD + n;
  const int iBase = iBlk * 64;
  const int tid  = threadIdx.x;
  const int lane = tid & 31, w = tid >> 5;
  const int lm = lane & 15, hi = lane >> 4;

  // Qw tile: async memory->LDS
  const unsigned qwBase = lds_addr32(qw_s);
  for (int c = tid; c < 512; c += 256) {
    int r = c >> 3, dof = (c & 7) << 3;
    unsigned ldsoff = qwBase + (unsigned)(r * 64 + dof) * 2u;
    unsigned goff   = (((unsigned)bh * S_LEN + (unsigned)(iBase + r)) * DH + (unsigned)dof) * 2u;
    async_ld_b128(ldsoff, goff, Qw);
  }
  // Qr tile: 80 rows with zero-fill past S (needs predicated zero writes)
  for (int c = tid; c < 640; c += 256) {
    int r = c >> 3, dof = (c & 7) << 3;
    int gi = iBase + r;
    uint4 v = make_uint4(0u, 0u, 0u, 0u);
    if (gi < S_LEN)
      v = *(const uint4*)&Qr[((size_t)bh * S_LEN + gi) * DH + dof];
    *(uint4*)&qr_s[r * 64 + dof] = v;
  }
  wait_async0();
  __syncthreads();

  // ---- AC = Qw * K^T : fill score slab ----
  for (int jt = w; jt < 64; jt += 8) {
    FragBF b0, b1;
    const __bf16* kp = Kh + ((size_t)bh * S_LEN + jt * 16 + lm) * DH;
    b0.u[0] = *(const uint4*)(kp + hi * 8);
    b0.u[1] = *(const uint4*)(kp + 16 + hi * 8);
    b1.u[0] = *(const uint4*)(kp + 32 + hi * 8);
    b1.u[1] = *(const uint4*)(kp + 48 + hi * 8);
    for (int it = 0; it < 4; ++it) {
      v8f acc = (v8f){0.f,0.f,0.f,0.f,0.f,0.f,0.f,0.f};
      FragBF a0, a1;
      const __bf16* qp = &qw_s[(it * 16 + lm) * 64];
      a0.u[0] = *(const uint4*)(qp + hi * 8);
      a0.u[1] = *(const uint4*)(qp + 16 + hi * 8);
      a1.u[0] = *(const uint4*)(qp + 32 + hi * 8);
      a1.u[1] = *(const uint4*)(qp + 48 + hi * 8);
      acc = wmma_bf16(a0.v, b0.v, acc);
      acc = wmma_bf16(a1.v, b1.v, acc);
      float* srow = scores + (it * 16 + hi * 8) * 1024 + jt * 16 + lm;
#pragma unroll
      for (int r = 0; r < 8; ++r) srow[r * 1024] = acc[r];
    }
  }
  __syncthreads();

  // ---- BD = Qr * R^T with EXACT rel-shift scatter-add ----
  // raw (i0, c): if c >= S-1-i0 -> out(i0, c-(S-1)+i0)  [lower band]
  //              else           -> out(i0-1, c+i0+1)    [reference's wrap region]
  for (int ct = w; ct < 64; ct += 8) {
    FragBF b0, b1;
    const __bf16* rp = Rh + ((size_t)n * S_LEN + ct * 16 + lm) * DH;
    b0.u[0] = *(const uint4*)(rp + hi * 8);
    b0.u[1] = *(const uint4*)(rp + 16 + hi * 8);
    b1.u[0] = *(const uint4*)(rp + 32 + hi * 8);
    b1.u[1] = *(const uint4*)(rp + 48 + hi * 8);
    for (int it2 = 0; it2 < 5; ++it2) {
      v8f acc = (v8f){0.f,0.f,0.f,0.f,0.f,0.f,0.f,0.f};
      FragBF a0, a1;
      const __bf16* qp = &qr_s[(it2 * 16 + lm) * 64];
      a0.u[0] = *(const uint4*)(qp + hi * 8);
      a0.u[1] = *(const uint4*)(qp + 16 + hi * 8);
      a1.u[0] = *(const uint4*)(qp + 32 + hi * 8);
      a1.u[1] = *(const uint4*)(qp + 48 + hi * 8);
      acc = wmma_bf16(a0.v, b0.v, acc);
      acc = wmma_bf16(a1.v, b1.v, acc);
#pragma unroll
      for (int r = 0; r < 8; ++r) {
        int i0 = iBase + it2 * 16 + hi * 8 + r;
        int cc = ct * 16 + lm;
        if (i0 < S_LEN) {
          int irow, j;
          if (cc >= S_LEN - 1 - i0) { irow = i0;     j = cc - (S_LEN - 1) + i0; }
          else                      { irow = i0 - 1; j = cc + i0 + 1; }
          int il = irow - iBase;
          if (il >= 0 && il < 64 && j >= 0 && j < S_LEN)
            scores[il * 1024 + j] += acc[r];   // bijective: no address collisions
        }
      }
    }
  }
  __syncthreads();

  // ---- softmax over keys (axis j), scale = 1/sqrt(DH) ----
  {
    const float scale = 0.125f;
    int row = tid >> 2, part = tid & 3;
    float* srow = scores + row * 1024 + part * 256;
    float m = -3.4e38f;
    for (int c = 0; c < 256; ++c) m = fmaxf(m, srow[c]);
    pbuf[tid] = m;
    __syncthreads();
    float rm = fmaxf(fmaxf(pbuf[row * 4], pbuf[row * 4 + 1]),
                     fmaxf(pbuf[row * 4 + 2], pbuf[row * 4 + 3]));
    float sum = 0.f;
    for (int c = 0; c < 256; ++c) {
      float e = __expf(scale * (srow[c] - rm));
      srow[c] = e;
      sum += e;
    }
    __syncthreads();
    pbuf[tid] = sum;
    __syncthreads();
    if (part == 0)
      rowsum[row] = pbuf[row * 4] + pbuf[row * 4 + 1] +
                    pbuf[row * 4 + 2] + pbuf[row * 4 + 3];
    __syncthreads();
  }

  // ---- O = P * V (unnormalized), divide by row sum at the end ----
  for (int tt = w * 2; tt < w * 2 + 2; ++tt) {
    int it = tt >> 2, dt = tt & 3;
    v8f acc = (v8f){0.f,0.f,0.f,0.f,0.f,0.f,0.f,0.f};
    for (int kt = 0; kt < 32; ++kt) {
      FragBF a, bb;
      __bf16* ae = reinterpret_cast<__bf16*>(&a);
      const float* srow = scores + (it * 16 + lm) * 1024 + kt * 32;
#pragma unroll
      for (int e = 0; e < 8; ++e) ae[e] = (__bf16)srow[hi * 8 + e];
#pragma unroll
      for (int e = 0; e < 8; ++e) ae[8 + e] = (__bf16)srow[16 + hi * 8 + e];
      const __bf16* vp = Vt + ((size_t)bh * DH + dt * 16 + lm) * S_LEN + kt * 32;
      bb.u[0] = *(const uint4*)(vp + hi * 8);
      bb.u[1] = *(const uint4*)(vp + 16 + hi * 8);
      acc = wmma_bf16(a.v, bb.v, acc);
    }
#pragma unroll
    for (int r = 0; r < 8; ++r) {
      int iLoc = it * 16 + hi * 8 + r;
      float val = acc[r] / rowsum[iLoc];
      int iGlob = iBase + iLoc;
      AV[((size_t)iGlob * BSZ + b) * NDIM + n * DH + dt * 16 + lm] = (__bf16)val;
    }
  }
}

// ---------------- residual + LayerNorm ----------------
__global__ __launch_bounds__(256) void add_ln_kernel(
    const float* __restrict__ hs, const float* __restrict__ attn,
    const float* __restrict__ gamma, const float* __restrict__ beta,
    float* __restrict__ out) {
  __shared__ float red[256];
  int row = blockIdx.x;
  int tid = threadIdx.x;
  float x[4];
  float s = 0.f;
#pragma unroll
  for (int e = 0; e < 4; ++e) {
    int c = tid * 4 + e;
    x[e] = hs[(size_t)row * DM + c] + attn[(size_t)row * DM + c];
    s += x[e];
  }
  red[tid] = s;
  __syncthreads();
  for (int off = 128; off > 0; off >>= 1) {
    if (tid < off) red[tid] += red[tid + off];
    __syncthreads();
  }
  float mu = red[0] * (1.f / 1024.f);
  __syncthreads();
  float v = 0.f;
#pragma unroll
  for (int e = 0; e < 4; ++e) { float d = x[e] - mu; v += d * d; }
  red[tid] = v;
  __syncthreads();
  for (int off = 128; off > 0; off >>= 1) {
    if (tid < off) red[tid] += red[tid + off];
    __syncthreads();
  }
  float rstd = rsqrtf(red[0] * (1.f / 1024.f) + 1e-5f);
#pragma unroll
  for (int e = 0; e < 4; ++e) {
    int c = tid * 4 + e;
    out[(size_t)row * DM + c] = (x[e] - mu) * rstd * gamma[c] + beta[c];
  }
}

// ---------------- launcher ----------------
extern "C" void kernel_launch(void* const* d_in, const int* in_sizes, int n_in,
                              void* d_out, int out_size, void* d_ws, size_t ws_size,
                              hipStream_t stream) {
  (void)in_sizes; (void)n_in; (void)out_size; (void)ws_size;
  const float* hs    = (const float*)d_in[0];
  const float* rel   = (const float*)d_in[1];
  const float* enc   = (const float*)d_in[2];
  const float* W_q   = (const float*)d_in[3];
  const float* W_kv  = (const float*)d_in[4];
  const float* W_r   = (const float*)d_in[5];
  const float* W_o   = (const float*)d_in[6];
  const float* rwb   = (const float*)d_in[7];
  const float* rrb   = (const float*)d_in[8];
  const float* gamma = (const float*)d_in[9];
  const float* beta  = (const float*)d_in[10];
  float* out = (float*)d_out;

  char* ws = (char*)d_ws;
  size_t off = 0;
  auto alloc = [&](size_t bytes) -> void* {
    void* p = ws + off;
    off += (bytes + 255) & ~(size_t)255;
    return p;
  };
  __bf16* hsb  = (__bf16*)alloc((size_t)SB * DM * 2);
  __bf16* encb = (__bf16*)alloc((size_t)SB * DM * 2);
  __bf16* relb = (__bf16*)alloc((size_t)S_LEN * DM * 2);
  __bf16* wqb  = (__bf16*)alloc((size_t)DM * NDIM * 2);
  __bf16* wkvb = (__bf16*)alloc((size_t)DM * 2 * NDIM * 2);
  __bf16* wrb  = (__bf16*)alloc((size_t)DM * NDIM * 2);
  __bf16* wob  = (__bf16*)alloc((size_t)NDIM * DM * 2);
  __bf16* Qb   = (__bf16*)alloc((size_t)SB * NDIM * 2);
  __bf16* KVb  = (__bf16*)alloc((size_t)SB * 2 * NDIM * 2);
  __bf16* Rb   = (__bf16*)alloc((size_t)S_LEN * NDIM * 2);
  __bf16* Qw   = (__bf16*)alloc((size_t)BH * S_LEN * DH * 2);
  __bf16* Qr   = (__bf16*)alloc((size_t)BH * S_LEN * DH * 2);
  __bf16* Kh   = (__bf16*)alloc((size_t)BH * S_LEN * DH * 2);
  __bf16* Vt   = (__bf16*)alloc((size_t)BH * DH * S_LEN * 2);
  __bf16* Rh   = (__bf16*)alloc((size_t)NHD * S_LEN * DH * 2);
  __bf16* AVb  = (__bf16*)alloc((size_t)SB * NDIM * 2);
  float*  Of   = (float*)alloc((size_t)SB * DM * 4);

  const int T = 256;
  cast_f32_bf16_kernel<<<4096, T, 0, stream>>>(hs,   hsb,  SB * DM);
  cast_f32_bf16_kernel<<<4096, T, 0, stream>>>(enc,  encb, SB * DM);
  cast_f32_bf16_kernel<<<1024, T, 0, stream>>>(rel,  relb, S_LEN * DM);
  cast_f32_bf16_kernel<<<1024, T, 0, stream>>>(W_q,  wqb,  DM * NDIM);
  cast_f32_bf16_kernel<<<2048, T, 0, stream>>>(W_kv, wkvb, DM * 2 * NDIM);
  cast_f32_bf16_kernel<<<1024, T, 0, stream>>>(W_r,  wrb,  DM * NDIM);
  cast_f32_bf16_kernel<<<1024, T, 0, stream>>>(W_o,  wob,  NDIM * DM);

  gemm_bf16_kernel<true><<<dim3(NDIM / 128, SB / 128), T, 0, stream>>>(
      hsb, wqb, (void*)Qb, SB, NDIM, DM);
  gemm_bf16_kernel<true><<<dim3(2 * NDIM / 128, SB / 128), T, 0, stream>>>(
      encb, wkvb, (void*)KVb, SB, 2 * NDIM, DM);
  gemm_bf16_kernel<true><<<dim3(NDIM / 128, S_LEN / 128), T, 0, stream>>>(
      relb, wrb, (void*)Rb, S_LEN, NDIM, DM);

  make_q_heads_kernel<<<(SB * NDIM) / 256, T, 0, stream>>>(Qb, rwb, rrb, Qw, Qr);
  make_kv_heads_kernel<<<(SB * NDIM) / 256, T, 0, stream>>>(KVb, Kh, Vt);
  make_r_heads_kernel<<<(S_LEN * NDIM) / 256, T, 0, stream>>>(Rb, Rh);

  size_t smem = (size_t)64 * 1024 * 4 + 64 * 64 * 2 + 80 * 64 * 2 + 256 * 4 + 64 * 4;
  attn_kernel<<<dim3(S_LEN / 64, NHD, BSZ), T, smem, stream>>>(
      Qw, Qr, Kh, Vt, Rh, AVb);

  gemm_bf16_kernel<false><<<dim3(DM / 128, SB / 128), T, 0, stream>>>(
      AVb, wob, (void*)Of, SB, DM, NDIM);

  add_ln_kernel<<<SB, T, 0, stream>>>(hs, Of, gamma, beta, out);
}
